// TransformerBlock_63445256896625
// MI455X (gfx1250) — matrix operand order repaired and checked by reading, not compiled, not run on hardware
//
#include <hip/hip_runtime.h>
#include <stdint.h>

#define EMBED 1024
#define HEADS 16
#define HDIM  64
#define SEQ   2048
#define BATCH 2
#define TOK   (BATCH*SEQ)
#define FFD   4096

typedef __attribute__((ext_vector_type(16))) __bf16 v16bf;
typedef __attribute__((ext_vector_type(8)))  float  v8f;

union AF { v16bf v; uint32_t u[8]; };

__device__ __forceinline__ uint16_t f2bf(float f) {
  uint32_t u = __float_as_uint(f);
  u += 0x7fffu + ((u >> 16) & 1u);        // round-to-nearest-even
  return (uint16_t)(u >> 16);
}
__device__ __forceinline__ float bf2f(uint16_t h) {
  return __uint_as_float(((uint32_t)h) << 16);
}
// ISA 7.12.2: 16-bit A-matrix 16x32 K index for VGPR-halfword pair v, lane-half
__device__ __forceinline__ int akmap(int v, int half) { return 2*v + (v>=4?8:0) + (half?8:0); }
// 16-bit B-matrix 32x16 K index
__device__ __forceinline__ int bkmap(int v, int half) { return 2*v + (v>=4?8:0) + (half?8:0); }

// Async global->LDS 16B copy per lane (CDNA5, tracked with ASYNCcnt).
// Generic pointers into LDS carry the LDS byte offset in the low 32 bits
// (ISA 10.2 aperture rules), so truncation yields the DS address.
__device__ __forceinline__ void async_copy_b128(const void* gptr, void* lptr) {
  uint32_t laddr = (uint32_t)(uintptr_t)lptr;
  uint64_t gaddr = (uint64_t)(uintptr_t)gptr;
  asm volatile("global_load_async_to_lds_b128 %0, %1, off"
               :: "v"(laddr), "v"(gaddr)
               : "memory");
}
__device__ __forceinline__ void wait_async0() {
  asm volatile("s_wait_asynccnt 0x0" ::: "memory");
}

// ---------------------------------------------------------------- converts
__global__ __launch_bounds__(256) void cvt_bf16_kernel(const float* __restrict__ in,
                                                       uint16_t* __restrict__ out, int n) {
  int i = (blockIdx.x * 256 + threadIdx.x) * 4;
  if (i + 3 < n) {
    float4 f = *(const float4*)(in + i);
    ushort4 o;
    o.x = f2bf(f.x); o.y = f2bf(f.y); o.z = f2bf(f.z); o.w = f2bf(f.w);
    *(ushort4*)(out + i) = o;
  }
}

// ---------------------------------------------------------------- layernorm
__global__ __launch_bounds__(256) void ln_kernel(const float* __restrict__ x,
                                                 const float* __restrict__ g,
                                                 const float* __restrict__ be,
                                                 uint16_t* __restrict__ out) {
  __shared__ float s_sum[256];
  __shared__ float s_sq[256];
  const int tid = threadIdx.x;
  const int row = blockIdx.x;
  const float* xr = x + (size_t)row * EMBED;
  float4 v = *(const float4*)(xr + tid * 4);
  s_sum[tid] = v.x + v.y + v.z + v.w;
  s_sq[tid]  = v.x*v.x + v.y*v.y + v.z*v.z + v.w*v.w;
  __syncthreads();
  for (int off = 128; off > 0; off >>= 1) {
    if (tid < off) { s_sum[tid] += s_sum[tid+off]; s_sq[tid] += s_sq[tid+off]; }
    __syncthreads();
  }
  const float mean = s_sum[0] * (1.0f / EMBED);
  const float var  = s_sq[0] * (1.0f / EMBED) - mean * mean;
  const float inv  = rsqrtf(var + 1e-5f);
  const int c = tid * 4;
  float4 gg = *(const float4*)(g + c);
  float4 bb = *(const float4*)(be + c);
  ushort4 o;
  o.x = f2bf((v.x - mean) * inv * gg.x + bb.x);
  o.y = f2bf((v.y - mean) * inv * gg.y + bb.y);
  o.z = f2bf((v.z - mean) * inv * gg.z + bb.z);
  o.w = f2bf((v.w - mean) * inv * gg.w + bb.w);
  *(ushort4*)(out + (size_t)row * EMBED + c) = o;
}

// ---------------------------------------------------------------- WMMA GEMM
// C[M,N] = A[M,K](bf16) * B[K,N](bf16), 128x128 tile / block of 8 waves.
// MODE 0: scatter to q/k/v [B,H,S,D], q scaled by 1/sqrt(HDIM)
// MODE 1: outF = C + bias + resid (f32)
// MODE 2: outH = bf16(relu(C + bias))
template<int MODE>
__global__ __launch_bounds__(256) void gemm_kernel(
    const uint16_t* __restrict__ A, const uint16_t* __restrict__ Bw,
    int M, int N, int K,
    const float* __restrict__ bias, const float* __restrict__ resid,
    float* __restrict__ outF, uint16_t* __restrict__ outH,
    uint16_t* __restrict__ qO, uint16_t* __restrict__ kO, uint16_t* __restrict__ vO) {
  __shared__ __align__(16) uint16_t As[128][40];   // [m][k], padded
  __shared__ __align__(16) uint16_t Bs[128][40];   // [n][k] (transposed), padded
  const int tid  = threadIdx.x;
  const int lane = tid & 31;
  const int w    = tid >> 5;
  const int half = lane >> 4;
  const int ln   = lane & 15;
  const int wm   = w >> 1;          // 0..3 (rows)
  const int wn   = w & 1;           // 0..1 (cols)
  const int m0   = blockIdx.y * 128;
  const int n0   = blockIdx.x * 128;

  v8f acc[2][4] = {};

  for (int kt = 0; kt < K; kt += 32) {
    // stage A: 128 rows x 32 k, async global->LDS (no VGPR round-trip)
#pragma unroll
    for (int i = 0; i < 2; i++) {
      int s = tid * 2 + i;                 // 0..511
      int r = s >> 2, kk = (s & 3) * 8;
      async_copy_b128(A + (size_t)(m0 + r) * K + kt + kk, &As[r][kk]);
    }
    // stage B transposed: 32 k x 128 n -> Bs[n][k]
#pragma unroll
    for (int i = 0; i < 2; i++) {
      int s = tid * 2 + i;
      int kr = s >> 4, nn = (s & 15) * 8;
      uint4 d = *(const uint4*)(Bw + (size_t)(kt + kr) * N + n0 + nn);
      const uint16_t* p = (const uint16_t*)&d;
#pragma unroll
      for (int j = 0; j < 8; j++) Bs[nn + j][kr] = p[j];
    }
    if (kt + 32 < K) {
      __builtin_prefetch(A  + (size_t)(m0 + (tid >> 1)) * K + kt + 32, 0, 1);
      __builtin_prefetch(Bw + (size_t)(kt + 32 + (tid >> 3)) * N + n0 + (tid & 7) * 16, 0, 1);
    }
    wait_async0();
    __syncthreads();

    AF af[2], bf[4];
#pragma unroll
    for (int mb = 0; mb < 2; mb++) {
      int mr = wm * 32 + mb * 16 + ln;
#pragma unroll
      for (int v2 = 0; v2 < 8; v2++)
        af[mb].u[v2] = *(const uint32_t*)&As[mr][akmap(v2, half)];
    }
#pragma unroll
    for (int nb = 0; nb < 4; nb++) {
      int nr = wn * 64 + nb * 16 + ln;
#pragma unroll
      for (int v2 = 0; v2 < 8; v2++)
        bf[nb].u[v2] = *(const uint32_t*)&Bs[nr][bkmap(v2, half)];
    }
#pragma unroll
    for (int mb = 0; mb < 2; mb++)
#pragma unroll
      for (int nb = 0; nb < 4; nb++)
        acc[mb][nb] = __builtin_amdgcn_wmma_f32_16x16x32_bf16(
            false, af[mb].v, false, bf[nb].v, (short)0, acc[mb][nb], false, false);
    __syncthreads();
  }

  // epilogue
#pragma unroll
  for (int mb = 0; mb < 2; mb++) {
#pragma unroll
    for (int nb = 0; nb < 4; nb++) {
#pragma unroll
      for (int r = 0; r < 8; r++) {
        int row = m0 + wm * 32 + mb * 16 + r + half * 8;
        int col = n0 + wn * 64 + nb * 16 + ln;
        float val = acc[mb][nb][r];
        if (MODE == 0) {
          int h = col / (3 * HDIM);
          int rem = col % (3 * HDIM);
          int t = rem / HDIM;
          int d = rem % HDIM;
          int b = row >> 11, s2 = row & (SEQ - 1);
          size_t dst = ((size_t)(b * HEADS + h) * SEQ + s2) * HDIM + d;
          if (t == 0)      qO[dst] = f2bf(val * 0.125f);   // fold 1/sqrt(64)
          else if (t == 1) kO[dst] = f2bf(val);
          else             vO[dst] = f2bf(val);
        } else if (MODE == 1) {
          size_t idx = (size_t)row * N + col;
          outF[idx] = val + bias[col] + resid[idx];
        } else {
          float vv = val + bias[col];
          outH[(size_t)row * N + col] = f2bf(vv > 0.0f ? vv : 0.0f);
        }
      }
    }
  }
}

// ---------------------------------------------------------------- flash attention
// grid (SEQ/64, BATCH*HEADS), 128 threads (4 waves x 16 q-rows)
__global__ __launch_bounds__(128) void attn_kernel(
    const uint16_t* __restrict__ qT, const uint16_t* __restrict__ kT,
    const uint16_t* __restrict__ vT, uint16_t* __restrict__ oT) {
  __shared__ __align__(16) uint16_t Ks[32][72];    // [kv][d]
  __shared__ __align__(16) uint16_t Vt[64][40];    // [d][kv] transposed
  __shared__ __align__(16) uint16_t Ps[4][16][40]; // per-wave P scratch [qrow][kv]

  const int tid  = threadIdx.x;
  const int lane = tid & 31, w = tid >> 5;
  const int half = lane >> 4, ln = lane & 15;
  const int bh   = blockIdx.y;
  const int qt   = blockIdx.x;
  const int h    = bh & (HEADS - 1), b = bh >> 4;
  const size_t base = (size_t)bh * SEQ * HDIM;
  const int qrow0 = qt * 64 + w * 16;

  // per-wave Q fragment (two K=32 chunks over D=64); scale folded in earlier
  AF aq[2];
  {
    int m = qrow0 + ln;
#pragma unroll
    for (int c = 0; c < 2; c++)
#pragma unroll
      for (int v2 = 0; v2 < 8; v2++)
        aq[c].u[v2] = *(const uint32_t*)(qT + base + (size_t)m * HDIM + c * 32 + akmap(v2, half));
  }

  float m_run[8], l_run[8];
#pragma unroll
  for (int r = 0; r < 8; r++) { m_run[r] = -1e30f; l_run[r] = 0.0f; }
  v8f Of[4] = {};

  const int kend = qt * 64 + 64;
  for (int kc = 0; kc < kend; kc += 32) {
    // stage K [32][64] async; V transposed [64][32] via VGPR path
#pragma unroll
    for (int i = 0; i < 2; i++) {
      int s = tid * 2 + i;               // 0..255
      int kr = s >> 3, dd = (s & 7) * 8;
      async_copy_b128(kT + base + (size_t)(kc + kr) * HDIM + dd, &Ks[kr][dd]);
      uint4 dv = *(const uint4*)(vT + base + (size_t)(kc + kr) * HDIM + dd);
      const uint16_t* p = (const uint16_t*)&dv;
#pragma unroll
      for (int j = 0; j < 8; j++) Vt[dd + j][kr] = p[j];
    }
    wait_async0();
    __syncthreads();

    const bool do0 = (kc      <= qrow0 + 15);
    const bool do1 = (kc + 16 <= qrow0 + 15);
    float alpha_prod[8];
#pragma unroll
    for (int r = 0; r < 8; r++) alpha_prod[r] = 1.0f;

#pragma unroll
    for (int kb = 0; kb < 2; kb++) {
      if ((kb == 0 && !do0) || (kb == 1 && !do1)) continue;
      // S = Q * K^T for a 16x16 block
      v8f sA = {};
#pragma unroll
      for (int c = 0; c < 2; c++) {
        AF bk;
#pragma unroll
        for (int v2 = 0; v2 < 8; v2++)
          bk.u[v2] = *(const uint32_t*)&Ks[kb * 16 + ln][c * 32 + bkmap(v2, half)];
        sA = __builtin_amdgcn_wmma_f32_16x16x32_bf16(
            false, aq[c].v, false, bk.v, (short)0, sA, false, false);
      }
      const int col = kc + kb * 16 + ln;
#pragma unroll
      for (int r = 0; r < 8; r++) {
        const int row = qrow0 + r + half * 8;
        float sv = (col <= row) ? sA[r] : -1e30f;    // causal mask
        float mx = sv;
        mx = fmaxf(mx, __shfl_xor(mx, 1));
        mx = fmaxf(mx, __shfl_xor(mx, 2));
        mx = fmaxf(mx, __shfl_xor(mx, 4));
        mx = fmaxf(mx, __shfl_xor(mx, 8));
        const float mnew  = fmaxf(m_run[r], mx);
        const float alpha = __expf(m_run[r] - mnew);
        const float p     = __expf(sv - mnew);
        float ps = p;
        ps += __shfl_xor(ps, 1);
        ps += __shfl_xor(ps, 2);
        ps += __shfl_xor(ps, 4);
        ps += __shfl_xor(ps, 8);
        l_run[r] = l_run[r] * alpha + ps;
        m_run[r] = mnew;
        alpha_prod[r] *= alpha;
        if (kb == 1) {   // re-normalize kb0 P entries to the new running max
          uint16_t* pp = &Ps[w][r + half * 8][ln];
          *pp = f2bf(bf2f(*pp) * alpha);
        }
        Ps[w][r + half * 8][kb * 16 + ln] = f2bf(p);
      }
    }

    if (do0) {
      if (!do1) {      // zero stale/unwritten upper 16 kv columns
#pragma unroll
        for (int r = 0; r < 8; r++) Ps[w][r + half * 8][16 + ln] = 0;
      }
      // O = diag(alpha)*O + P * V
      AF pf;
#pragma unroll
      for (int v2 = 0; v2 < 8; v2++)
        pf.u[v2] = *(const uint32_t*)&Ps[w][ln][akmap(v2, half)];
#pragma unroll
      for (int nb = 0; nb < 4; nb++) {
        AF vf;
#pragma unroll
        for (int v2 = 0; v2 < 8; v2++)
          vf.u[v2] = *(const uint32_t*)&Vt[nb * 16 + ln][bkmap(v2, half)];
        v8f c2;
#pragma unroll
        for (int r = 0; r < 8; r++) c2[r] = Of[nb][r] * alpha_prod[r];
        Of[nb] = __builtin_amdgcn_wmma_f32_16x16x32_bf16(
            false, pf.v, false, vf.v, (short)0, c2, false, false);
      }
    }
    __syncthreads();
  }

  // normalize and write o[token][h*64+d]
  float inv_l[8];
#pragma unroll
  for (int r = 0; r < 8; r++) inv_l[r] = 1.0f / l_run[r];
#pragma unroll
  for (int nb = 0; nb < 4; nb++)
#pragma unroll
    for (int r = 0; r < 8; r++) {
      const int row = qrow0 + r + half * 8;
      const int d   = nb * 16 + ln;
      const size_t tok = (size_t)b * SEQ + row;
      oT[tok * (HEADS * HDIM) + h * HDIM + d] = f2bf(Of[nb][r] * inv_l[r]);
    }
}

// ---------------------------------------------------------------- launch
extern "C" void kernel_launch(void* const* d_in, const int* in_sizes, int n_in,
                              void* d_out, int out_size, void* d_ws, size_t ws_size,
                              hipStream_t stream) {
  const float* x      = (const float*)d_in[0];
  const float* w_qkv  = (const float*)d_in[1];
  const float* w_proj = (const float*)d_in[2];
  const float* b_proj = (const float*)d_in[3];
  const float* w1     = (const float*)d_in[4];
  const float* b1     = (const float*)d_in[5];
  const float* w2     = (const float*)d_in[6];
  const float* b2     = (const float*)d_in[7];
  const float* g1     = (const float*)d_in[8];
  const float* be1    = (const float*)d_in[9];
  const float* g2     = (const float*)d_in[10];
  const float* be2    = (const float*)d_in[11];
  float* out = (float*)d_out;

  char* ws = (char*)d_ws;
  size_t off = 0;
  auto alloc = [&](size_t bytes) -> void* {
    void* p = ws + off;
    off += (bytes + 255) & ~(size_t)255;
    return p;
  };
  uint16_t* wqkv_h  = (uint16_t*)alloc((size_t)EMBED * 3 * HEADS * HDIM * 2);
  uint16_t* wproj_h = (uint16_t*)alloc((size_t)EMBED * EMBED * 2);
  uint16_t* w1_h    = (uint16_t*)alloc((size_t)EMBED * FFD * 2);
  uint16_t* w2_h    = (uint16_t*)alloc((size_t)FFD * EMBED * 2);
  uint16_t* h_bf    = (uint16_t*)alloc((size_t)TOK * EMBED * 2);
  uint16_t* q_bf    = (uint16_t*)alloc((size_t)TOK * EMBED * 2);
  uint16_t* k_bf    = (uint16_t*)alloc((size_t)TOK * EMBED * 2);
  uint16_t* v_bf    = (uint16_t*)alloc((size_t)TOK * EMBED * 2);
  uint16_t* o_bf    = (uint16_t*)alloc((size_t)TOK * EMBED * 2);
  float*    x2      = (float*)   alloc((size_t)TOK * EMBED * 4);
  uint16_t* h2_bf   = (uint16_t*)alloc((size_t)TOK * EMBED * 2);
  uint16_t* ff_bf   = (uint16_t*)alloc((size_t)TOK * FFD * 2);

  // weight casts
  cvt_bf16_kernel<<<(EMBED * 3 * HEADS * HDIM) / 1024, 256, 0, stream>>>(w_qkv, wqkv_h, EMBED * 3 * HEADS * HDIM);
  cvt_bf16_kernel<<<(EMBED * EMBED) / 1024, 256, 0, stream>>>(w_proj, wproj_h, EMBED * EMBED);
  cvt_bf16_kernel<<<(EMBED * FFD) / 1024, 256, 0, stream>>>(w1, w1_h, EMBED * FFD);
  cvt_bf16_kernel<<<(FFD * EMBED) / 1024, 256, 0, stream>>>(w2, w2_h, FFD * EMBED);

  // ln1
  ln_kernel<<<TOK, 256, 0, stream>>>(x, g1, be1, h_bf);

  // qkv = h @ w_qkv  -> scatter q,k,v
  gemm_kernel<0><<<dim3((3 * EMBED) / 128, TOK / 128), 256, 0, stream>>>(
      h_bf, wqkv_h, TOK, 3 * EMBED, EMBED, nullptr, nullptr, nullptr, nullptr,
      q_bf, k_bf, v_bf);

  // flash attention
  attn_kernel<<<dim3(SEQ / 64, BATCH * HEADS), 128, 0, stream>>>(q_bf, k_bf, v_bf, o_bf);

  // x2 = x + o @ w_proj + b_proj
  gemm_kernel<1><<<dim3(EMBED / 128, TOK / 128), 256, 0, stream>>>(
      o_bf, wproj_h, TOK, EMBED, EMBED, b_proj, x, x2, nullptr,
      nullptr, nullptr, nullptr);

  // ln2
  ln_kernel<<<TOK, 256, 0, stream>>>(x2, g2, be2, h2_bf);

  // ff = relu(h2 @ w1 + b1)
  gemm_kernel<2><<<dim3(FFD / 128, TOK / 128), 256, 0, stream>>>(
      h2_bf, w1_h, TOK, FFD, EMBED, b1, nullptr, nullptr, ff_bf,
      nullptr, nullptr, nullptr);

  // out = x2 + ff @ w2 + b2
  gemm_kernel<1><<<dim3(EMBED / 128, TOK / 128), 256, 0, stream>>>(
      ff_bf, w2_h, TOK, EMBED, FFD, b2, x2, out, nullptr,
      nullptr, nullptr, nullptr);

  (void)in_sizes; (void)n_in; (void)out_size; (void)ws_size;
}
